// ContrastiveLoss_20323785244981
// MI455X (gfx1250) — matrix operand order repaired and checked by reading, not compiled, not run on hardware
//
#include <hip/hip_runtime.h>
#include <hip/hip_bf16.h>
#include <stdint.h>

#define N_EMB 8192
#define D_EMB 512
#define KC    64
#define NCHUNK (D_EMB / KC)   // 8
#define LDSW  72   // halfs per LDS row: 64 data + 8 pad (TDM pad: 16B per 128B row)

typedef __attribute__((ext_vector_type(16))) __bf16 v16bf;
typedef __attribute__((ext_vector_type(8)))  __bf16 bf16x8;
typedef __attribute__((ext_vector_type(8)))  float  v8f;
typedef __attribute__((ext_vector_type(4)))  unsigned int u32x4;
typedef __attribute__((ext_vector_type(8)))  int i32x8;
typedef __attribute__((ext_vector_type(4)))  int i32x4;

// ---------------------------------------------------------------------------
// TDM: load one 128(row) x 64(k) bf16 tile (tensor stride 512 halfs) into LDS,
// padding LDS dest by 4 DWORDs (16B) after every 32 DWORDs (one 128B row)
// => LDS row stride 144B = 72 halfs.  D# built per CDNA5 ISA.
// ---------------------------------------------------------------------------
__device__ __forceinline__ void tdm_load_tile(unsigned ldsAddr, unsigned long long gAddr) {
  u32x4 g0;
  g0[0] = 1u;                                        // count=1, user descriptor
  g0[1] = ldsAddr;                                   // lds_addr (bytes)
  g0[2] = (unsigned)gAddr;                           // global_addr[31:0]
  g0[3] = ((unsigned)(gAddr >> 32) & 0x01FFFFFFu)    // global_addr[56:32]
        | (2u << 30);                                // type=2 ("image")
  i32x8 g1;
  g1[0] = (int)((1u << 16)      // data_size = 1 -> 2 bytes
              | (1u << 20)      // pad_enable
              | (4u << 22)      // pad_interval: 32 DWORDs (=128B row)
              | (3u << 25));    // pad_amount: 4 DWORDs (=16B)
  g1[1] = (int)(512u << 16);    // tensor_dim0[15:0]=512 -> w1[31:16]
  g1[2] = (int)(8192u << 16);   // tensor_dim0 hi=0, tensor_dim1[15:0]=8192
  g1[3] = (int)(64u << 16);     // tensor_dim1 hi=0, tile_dim0=64 (k chunk)
  g1[4] = 128;                  // tile_dim1=128 rows, tile_dim2=0
  g1[5] = 512;                  // tensor_dim0_stride low = 512
  g1[6] = 0;                    // stride hi / dim1_stride
  g1[7] = 0;
  i32x4 z4 = {0, 0, 0, 0};
#if __has_builtin(__builtin_amdgcn_tensor_load_to_lds)
  #if __clang_major__ >= 23
  i32x8 z8 = {0, 0, 0, 0, 0, 0, 0, 0};
  __builtin_amdgcn_tensor_load_to_lds(g0, g1, z4, z4, z8, 0);
  #else
  __builtin_amdgcn_tensor_load_to_lds(g0, g1, z4, z4, 0);
  #endif
#else
  asm volatile("tensor_load_to_lds %0, %1" :: "s"(g0), "s"(g1) : "memory");
#endif
}

// ---------------------------------------------------------------------------
// Kernel 1: fp32 -> bf16 conversion + zero accumulators/output
// ---------------------------------------------------------------------------
__global__ void __launch_bounds__(256) cl_convert(const float* __restrict__ in,
                                                  __bf16* __restrict__ outB,
                                                  float* __restrict__ accum,
                                                  float* __restrict__ lossOut) {
  const int gid = blockIdx.x * 256 + threadIdx.x;     // 524288 threads, 8 elems each
  const float4* in4 = (const float4*)in;
  float4 a = in4[gid * 2 + 0];
  float4 b = in4[gid * 2 + 1];
  bf16x8 o;
  o[0] = (__bf16)a.x; o[1] = (__bf16)a.y; o[2] = (__bf16)a.z; o[3] = (__bf16)a.w;
  o[4] = (__bf16)b.x; o[5] = (__bf16)b.y; o[6] = (__bf16)b.z; o[7] = (__bf16)b.w;
  ((bf16x8*)outB)[gid] = o;
  if (gid < 2 * N_EMB) accum[gid] = 0.0f;             // gPos[8192] ++ gTot[8192]
  if (gid == 0) lossOut[0] = 0.0f;
}

// ---------------------------------------------------------------------------
// Kernel 2: fused bf16 WMMA GEMM (sim = E E^T / T) + exp/mask/row-sum epilogue
// Grid (64,64) of 128x128 sim tiles; 256 threads = 8 waves in a 2x4 grid,
// each wave owns 64 rows x 32 cols = 4x2 wmma accumulators.
// TDM loads are double-buffered: chunk c+1 DMA overlaps chunk c compute.
// ---------------------------------------------------------------------------
__global__ void __launch_bounds__(256) cl_gemm(const __bf16* __restrict__ emb,
                                               const long long* __restrict__ labels,
                                               float* __restrict__ gPos,
                                               float* __restrict__ gTot) {
  __shared__ __align__(16) __bf16 sA[2][128 * LDSW];
  __shared__ __align__(16) __bf16 sB[2][128 * LDSW];
  __shared__ int sLabR[128];
  __shared__ int sLabC[128];

  const int bx = blockIdx.x, by = blockIdx.y;
  const int tid = threadIdx.x;
  if (tid < 128) sLabR[tid] = (int)labels[by * 128 + tid];
  else           sLabC[tid - 128] = (int)labels[bx * 128 + (tid - 128)];

  const int wave = tid >> 5, lane = tid & 31;
  const int waveM = wave >> 2, waveN = wave & 3;     // 2 x 4 wave grid
  const int l16 = lane & 15, lhalf = lane >> 4;

  v8f acc[4][2] = {};

  const unsigned ldsA0 = (unsigned)(uintptr_t)(void*)sA[0];
  const unsigned ldsA1 = (unsigned)(uintptr_t)(void*)sA[1];
  const unsigned ldsB0 = (unsigned)(uintptr_t)(void*)sB[0];
  const unsigned ldsB1 = (unsigned)(uintptr_t)(void*)sB[1];
  const unsigned long long gA =
      (unsigned long long)(uintptr_t)(emb + (size_t)by * 128 * D_EMB);
  const unsigned long long gB =
      (unsigned long long)(uintptr_t)(emb + (size_t)bx * 128 * D_EMB);

  // prologue: kick off chunk 0 DMA (wave0 -> A tile, wave1 -> B tile)
  if (wave == 0) tdm_load_tile(ldsA0, gA);
  if (wave == 1) tdm_load_tile(ldsB0, gB);

#pragma unroll
  for (int c = 0; c < NCHUNK; ++c) {
    const int cur = c & 1;
    // prefetch next chunk into the other buffer (its last readers finished
    // at the barrier that ended chunk c-1)
    if (c + 1 < NCHUNK) {
      const unsigned long long off = (unsigned long long)(c + 1) * KC * 2;
      if (wave == 0) tdm_load_tile(cur ? ldsA0 : ldsA1, gA + off);
      if (wave == 1) tdm_load_tile(cur ? ldsB0 : ldsB1, gB + off);
      __builtin_amdgcn_s_wait_tensorcnt(1);   // chunk c done; c+1 may remain
    } else {
      __builtin_amdgcn_s_wait_tensorcnt(0);   // last chunk: drain
    }
    __syncthreads();                          // tiles visible to all waves

    const __bf16* tA = sA[cur];
    const __bf16* tB = sB[cur];
#pragma unroll
    for (int ks = 0; ks < KC; ks += 32) {
      v16bf aF[4], bF[2];
#pragma unroll
      for (int m = 0; m < 4; ++m) {
        // A frag (16x32): lane holds row m=l16; halfs 0..7 = K koff..koff+7,
        // halfs 8..15 = K koff+16..+23, koff = ks + 8*lhalf  (ISA 7.12.2)
        const __bf16* p = &tA[(waveM * 64 + m * 16 + l16) * LDSW + ks + lhalf * 8];
        bf16x8 lo = *(const bf16x8*)p;
        bf16x8 hi = *(const bf16x8*)(p + 16);
        aF[m] = __builtin_shufflevector(lo, hi, 0,1,2,3,4,5,6,7,8,9,10,11,12,13,14,15);
      }
#pragma unroll
      for (int n = 0; n < 2; ++n) {
        // B frag (32x16): lane holds col n=l16; halfs 0..7 = K kb..kb+7,
        // halfs 8..15 = K kb+16..+23, kb = ks + 8*lhalf   (B col n of sim tile == row n of E)
        const __bf16* p = &tB[(waveN * 32 + n * 16 + l16) * LDSW + ks + lhalf * 8];
        bf16x8 lo = *(const bf16x8*)p;
        bf16x8 hi = *(const bf16x8*)(p + 16);
        bF[n] = __builtin_shufflevector(lo, hi, 0,1,2,3,4,5,6,7,8,9,10,11,12,13,14,15);
      }
#pragma unroll
      for (int m = 0; m < 4; ++m)
#pragma unroll
        for (int n = 0; n < 2; ++n)
          acc[m][n] = __builtin_amdgcn_wmma_f32_16x16x32_bf16(
              false, aF[m], false, bF[n], (short)0, acc[m][n], false, false);
    }
    __syncthreads();                          // buffer free for next prefetch
  }

  // ---- fused epilogue: exp(sim/T), exclude diagonal, pos/total row sums ----
  const float invT = 1.0f / 0.07f;
#pragma unroll
  for (int m = 0; m < 4; ++m) {
#pragma unroll
    for (int v = 0; v < 8; ++v) {
      const int rloc = waveM * 64 + m * 16 + v + lhalf * 8;  // C layout: M=v(+8)
      const int row  = by * 128 + rloc;
      const int lr   = sLabR[rloc];
      float p = 0.0f, t = 0.0f;
#pragma unroll
      for (int n = 0; n < 2; ++n) {
        const int cloc = waveN * 32 + n * 16 + l16;          // C layout: N=lane
        const int col  = bx * 128 + cloc;
        const float e  = __expf(acc[m][n][v] * invT);
        if (row != col) {
          t += e;
          if (lr == sLabC[cloc]) p += e;
        }
      }
      // reduce across the 16 lanes holding this row's columns
#pragma unroll
      for (int d = 1; d < 16; d <<= 1) {
        p += __shfl_xor(p, d, 16);
        t += __shfl_xor(t, d, 16);
      }
      if (l16 == 0) {
        atomicAdd(&gPos[row], p);
        atomicAdd(&gTot[row], t);
      }
    }
  }
}

// ---------------------------------------------------------------------------
// Kernel 3: per-row loss + mean reduction
// ---------------------------------------------------------------------------
__global__ void __launch_bounds__(256) cl_finalize(const float* __restrict__ gPos,
                                                   const float* __restrict__ gTot,
                                                   float* __restrict__ out) {
  const int i = blockIdx.x * 256 + threadIdx.x;       // 32 blocks * 256 = 8192
  const float p = gPos[i], t = gTot[i];
  float loss = -__logf(p / (t + 1e-8f) + 1e-8f);
#pragma unroll
  for (int d = 1; d < 32; d <<= 1) loss += __shfl_xor(loss, d, 32);
  __shared__ float red[8];
  if ((threadIdx.x & 31) == 0) red[threadIdx.x >> 5] = loss;
  __syncthreads();
  if (threadIdx.x == 0) {
    float s = 0.0f;
#pragma unroll
    for (int w = 0; w < 8; ++w) s += red[w];
    atomicAdd(out, s * (1.0f / N_EMB));
  }
}

// ---------------------------------------------------------------------------
extern "C" void kernel_launch(void* const* d_in, const int* in_sizes, int n_in,
                              void* d_out, int out_size, void* d_ws, size_t ws_size,
                              hipStream_t stream) {
  const float*     emb    = (const float*)d_in[0];       // [8192, 512] fp32
  const long long* labels = (const long long*)d_in[1];   // [8192] int64
  float* out = (float*)d_out;

  char*   ws   = (char*)d_ws;
  __bf16* embB = (__bf16*)ws;                                   // 8 MB
  float*  gPos = (float*)(ws + (size_t)N_EMB * D_EMB * 2);      // 32 KB
  float*  gTot = gPos + N_EMB;                                  // 32 KB

  // 1) convert fp32->bf16, zero accumulators & output
  cl_convert<<<(N_EMB * D_EMB / 8) / 256, 256, 0, stream>>>(emb, embB, gPos, out);

  // 2) fused WMMA GEMM + epilogue over 64x64 tiles of the 8192x8192 sim matrix
  dim3 grid(N_EMB / 128, N_EMB / 128);
  cl_gemm<<<grid, 256, 0, stream>>>(embB, labels, gPos, gTot);

  // 3) loss + mean
  cl_finalize<<<N_EMB / 256, 256, 0, stream>>>(gPos, gTot, out);
}
